// ChebLayer_17703855194471
// MI455X (gfx1250) — matrix operand order repaired and checked by reading, not compiled, not run on hardware
//
#include <hip/hip_runtime.h>
#include <hip/hip_bf16.h>

typedef __attribute__((ext_vector_type(16))) __bf16 v16bf;
typedef __attribute__((ext_vector_type(8)))  __bf16 v8bf;
typedef __attribute__((ext_vector_type(8)))  float  v8f;

#define FDIM 128
#define KCHEB 4

// ---------------------------------------------------------------------------
// Pack weight[K,128,128] f32 -> bf16 in per-lane WMMA B-operand order.
// Contiguous k order used here (same as A), B (32x16):
//   lane = n + 16*(K>=16-half), element e = K%16  (K half selected by lane>=16)
// Packed index: (((k*8+ct)*4+kc)*32 + lane)*16 + e
// ---------------------------------------------------------------------------
__global__ void pack_weights(const float* __restrict__ W, __bf16* __restrict__ packed) {
    int t = blockIdx.x * blockDim.x + threadIdx.x;
    if (t >= KCHEB * 8 * 4 * 32) return;
    int lane = t & 31;
    int kc   = (t >> 5) & 3;   // K-chunk of 32
    int ct   = (t >> 7) & 7;   // 16-col tile
    int k    = (t >> 10) & 3;  // chebyshev term
    int n     = lane & 15;
    int khalf = (lane >> 4) * 16;
    __bf16* dst = packed + (size_t)t * 16;
#pragma unroll
    for (int e = 0; e < 16; ++e) {
        int kk = kc * 32 + khalf + e;
        dst[e] = (__bf16)W[((size_t)(k * FDIM + kk)) * FDIM + ct * 16 + n];
    }
}

// dst = alpha * src  (alpha=0 zero-init, alpha=-1 for "- T_{k-2}" term)
__global__ void scale_copy(const float* __restrict__ src, float* __restrict__ dst,
                           float alpha, size_t n) {
    size_t i = (size_t)blockIdx.x * blockDim.x + threadIdx.x;
    if (i < n) dst[i] = alpha * src[i];
}

// ---------------------------------------------------------------------------
// SpMM scatter: acc[dst[e]] += scale * edge_w[e] * h[src[e]]
// One wave32 per edge; 4 floats per lane covers 128 features.
// h (51 MB) is L2-resident (192 MB L2) -> gathers/atomics mostly hit L2.
// ---------------------------------------------------------------------------
__global__ void spmm_scatter(const float* __restrict__ h,
                             const int* __restrict__ src,
                             const int* __restrict__ dst,
                             const float* __restrict__ ew,
                             float scale, float* __restrict__ acc, int E) {
    int gwarp = (int)((blockIdx.x * (size_t)blockDim.x + threadIdx.x) >> 5);
    int lane  = threadIdx.x & 31;
    if (gwarp >= E) return;
    int s = src[gwarp];
    int d = dst[gwarp];
    float w = ew[gwarp] * scale;
    const float4* hv = (const float4*)(h + (size_t)s * FDIM);
    float4 v = hv[lane];
    float* a = acc + (size_t)d * FDIM + lane * 4;
    unsafeAtomicAdd(a + 0, w * v.x);
    unsafeAtomicAdd(a + 1, w * v.y);
    unsafeAtomicAdd(a + 2, w * v.z);
    unsafeAtomicAdd(a + 3, w * v.w);
}

// ---------------------------------------------------------------------------
// Fused epilogue: out[16-row tile] = sum_k (c_k ⊙ T_k) @ W_k + bias
// Block = 256 threads = 8 waves; wave w owns columns [16w,16w+16).
// A tiles (scaled T_k) staged to LDS as bf16; 16 v_wmma_f32_16x16x32_bf16/wave.
// ---------------------------------------------------------------------------
__global__ void cheb_gemm(const float* __restrict__ T0, const float* __restrict__ T1,
                          const float* __restrict__ T2, const float* __restrict__ T3,
                          const float* __restrict__ fc, const float* __restrict__ bias,
                          const __bf16* __restrict__ packedW,
                          float* __restrict__ out, int n) {
    __shared__ __bf16 As[KCHEB][16][FDIM];   // 16 KB

    const float* T[KCHEB] = {T0, T1, T2, T3};
    int row_base = blockIdx.x * 16;
    int tid = threadIdx.x;

    // Stage A: As[k][m][kk] = bf16( fc[k][row] * T_k[row][kk] )
    for (int k = 0; k < KCHEB; ++k) {
        const float* srcT = T[k];
        const float* fck  = fc + (size_t)k * n;
        for (int i = tid; i < 16 * FDIM; i += 256) {
            int m  = i >> 7;
            int kk = i & (FDIM - 1);
            int row = row_base + m;
            float v = 0.0f;
            if (row < n) v = fck[row] * srcT[(size_t)row * FDIM + kk];
            As[k][m][kk] = (__bf16)v;
        }
    }
    __syncthreads();

    int wave = tid >> 5;
    int lane = tid & 31;
    int nlo  = lane & 15;          // A: M index / B,D: N index
    bool hi  = lane >= 16;
    int col_base = wave * 16;

    float bval = bias[col_base + nlo];      // bias is per-column, uniform over M
    v8f acc;
#pragma unroll
    for (int r = 0; r < 8; ++r) acc[r] = bval;

    for (int k = 0; k < KCHEB; ++k) {
#pragma unroll
        for (int kc = 0; kc < 4; ++kc) {
            // A operand: lanes 0-15 take K = base+{0..15}; lanes 16-31 +16
            int abase = kc * 32 + (hi ? 16 : 0);
            v8bf a0 = *(const v8bf*)&As[k][nlo][abase];
            v8bf a1 = *(const v8bf*)&As[k][nlo][abase + 8];
            v16bf a;
#pragma unroll
            for (int e = 0; e < 8; ++e) { a[e] = a0[e]; a[8 + e] = a1[e]; }

            // B operand: contiguous 32B per lane from the pre-packed buffer
            const __bf16* bp = packedW +
                (size_t)((((k * 8 + wave) * 4) + kc) * 32 + lane) * 16;
            v16bf b = *(const v16bf*)bp;

            acc = __builtin_amdgcn_wmma_f32_16x16x32_bf16(
                false, a, false, b, (short)0, acc, false, false);
        }
    }

    // D layout: VGPR r -> M = r + 8*hi, N = col_base + nlo
#pragma unroll
    for (int r = 0; r < 8; ++r) {
        int row = row_base + r + (hi ? 8 : 0);
        if (row < n) out[(size_t)row * FDIM + col_base + nlo] = acc[r];
    }
}

extern "C" void kernel_launch(void* const* d_in, const int* in_sizes, int n_in,
                              void* d_out, int out_size, void* d_ws, size_t ws_size,
                              hipStream_t stream) {
    const float* x    = (const float*)d_in[0];
    const float* fc   = (const float*)d_in[1];
    const float* W    = (const float*)d_in[2];
    const float* bias = (const float*)d_in[3];
    const float* ew   = (const float*)d_in[4];
    const int*   src  = (const int*)d_in[5];
    const int*   dst  = (const int*)d_in[6];

    const int N = in_sizes[0] / FDIM;
    const int E = in_sizes[4];
    const size_t NF = (size_t)N * FDIM;

    float*  Tx1     = (float*)d_ws;
    float*  Tx2     = Tx1 + NF;
    float*  Tx3     = Tx2 + NF;
    __bf16* packedW = (__bf16*)(Tx3 + NF);   // 128 KB
    float*  out     = (float*)d_out;

    // Pre-pack bf16 weights in WMMA B-operand lane order (once per launch).
    pack_weights<<<16, 256, 0, stream>>>(W, packedW);

    int eltBlocks  = (int)((NF + 255) / 256);
    int spmmBlocks = (E + 7) / 8;            // 8 edges (waves) per 256-thr block

    // T1 = L x           : acc starts at 0
    scale_copy<<<eltBlocks, 256, 0, stream>>>(x, Tx1, 0.0f, NF);
    // T2 = 2 L T1 - T0   : acc starts at -x, scatter with scale 2
    scale_copy<<<eltBlocks, 256, 0, stream>>>(x, Tx2, -1.0f, NF);
    spmm_scatter<<<spmmBlocks, 256, 0, stream>>>(x, src, dst, ew, 1.0f, Tx1, E);
    // T3 = 2 L T2 - T1   : acc starts at -T1 (after T1 is final)
    scale_copy<<<eltBlocks, 256, 0, stream>>>(Tx1, Tx3, -1.0f, NF);
    spmm_scatter<<<spmmBlocks, 256, 0, stream>>>(Tx1, src, dst, ew, 2.0f, Tx2, E);
    spmm_scatter<<<spmmBlocks, 256, 0, stream>>>(Tx2, src, dst, ew, 2.0f, Tx3, E);

    // Fused scale + 4-way GEMM + bias via bf16 WMMA, f32 accumulate.
    cheb_gemm<<<(N + 15) / 16, 256, 0, stream>>>(x, Tx1, Tx2, Tx3, fc, bias,
                                                 packedW, out, N);
}
